// ForwardBackwardGNN_47081431499229
// MI455X (gfx1250) — matrix operand order repaired and checked by reading, not compiled, not run on hardware
//
#include <hip/hip_runtime.h>
#include <hip/hip_bf16.h>

typedef _Float16 f16;
typedef __attribute__((ext_vector_type(16))) _Float16 v16h;
typedef __attribute__((ext_vector_type(8)))  _Float16 v8h;
typedef __attribute__((ext_vector_type(8)))  float    v8f;

// global (address_space(1)) views — force global_load/store instead of flat_*
typedef f16      __attribute__((address_space(1))) gf16;
typedef float    __attribute__((address_space(1))) gf32;
typedef int      __attribute__((address_space(1))) gi32;
typedef unsigned __attribute__((address_space(1))) gu32;
typedef v16h     __attribute__((address_space(1))) g_v16h;
typedef v8h      __attribute__((address_space(1))) g_v8h;

// ---------------- problem constants ----------------
constexpr int N_NODES = 100000;
constexpr int N_EDGES = 400000;
constexpr int SEQ   = 16;
constexpr int VOCAB = 128;
constexpr int EDIM  = 32;   // embedding dim (LSTM input)
constexpr int HDIM  = 64;   // LSTM hidden
constexpr int G4H   = 256;  // 4*HDIM gate width
constexpr int HID   = 214;  // node feature dim
constexpr int HP    = 224;  // padded to 14*16 for WMMA
constexpr int OUTC  = 428;  // output columns (two halves)
constexpr int STATE = 53;
constexpr int REGEX = 161;
constexpr float NEG = 0.2f;

// ---------------- workspace layout (bytes) ----------------
constexpr size_t OFF_EMB    = 0;                         // 128*32 f16     = 8192
constexpr size_t OFF_WIH_F  = 8192;                      // 256*32 f16     = 16384
constexpr size_t OFF_WHH_F  = 24576;                     // 256*64 f16     = 32768
constexpr size_t OFF_WIH_B  = 57344;                     // 16384
constexpr size_t OFF_WHH_B  = 73728;                     // 32768
constexpr size_t OFF_W1     = 106496;                    // 32*128 f16     = 8192
constexpr size_t OFF_BIAS_F = 114688;                    // 256 f32        = 1024
constexpr size_t OFF_BIAS_B = 115712;                    // 1024
constexpr size_t OFF_WL     = 116736;                    // 224*224 f16    = 100352
constexpr size_t OFF_WR     = 217088;                    // 100352
constexpr size_t OFF_EA     = 317440;                    // E f32          = 1600000
constexpr size_t OFF_ER     = 1917440;                   // 1600000
constexpr size_t OFF_EX     = 3517440;                   // 1600000
constexpr size_t OFF_M      = 5117440;                   // N u32          = 400000
constexpr size_t OFF_DEN    = 5517568;                   // N f32          = 400000
constexpr size_t OFF_XH     = 5917696;                   // N*224 f16      = 44800000
constexpr size_t OFF_XL     = 50717696;                  // N*224 f32      = 89600000
constexpr size_t OFF_XR     = 140317696;                 // 89600000
constexpr size_t OFF_ACC    = 229917696;                 // 89600000  (end 319517696)

// ---------------- helpers ----------------
__device__ __forceinline__ float sigm(float x) { return 1.f / (1.f + __expf(-x)); }
// branchless tanh: 1 - 2/(exp(2x)+1); saturates to +/-1 correctly
__device__ __forceinline__ float tanh_fast(float x) {
  return 1.f - 2.f / (__expf(2.f * x) + 1.f);
}

__device__ __forceinline__ v16h cat16(v8h a, v8h b) {
  return __builtin_shufflevector(a, b, 0,1,2,3,4,5,6,7,8,9,10,11,12,13,14,15);
}

// order-preserving float<->uint encoding for atomicMax on floats
__device__ __forceinline__ unsigned f2ord(float f) {
  unsigned u = __float_as_uint(f);
  return (u & 0x80000000u) ? ~u : (u | 0x80000000u);
}
__device__ __forceinline__ float ord2f(unsigned u) {
  return (u & 0x80000000u) ? __uint_as_float(u & 0x7fffffffu) : __uint_as_float(~u);
}

// global-AS view of a pointer (inttoptr into AS1; no addrspacecast, no flat path)
__device__ __forceinline__ unsigned long long g_u64(const void* p) {
  return (unsigned long long)p;
}
// opaque variant: also defeats LICM hoist+spill of loop-invariant weight loads
__device__ __forceinline__ unsigned long long g_u64_opq(const void* p) {
  unsigned long long u = (unsigned long long)p;
  asm volatile("" : "+s"(u));
  return u;
}

#define WMMA_F16(A, B, C) \
  __builtin_amdgcn_wmma_f32_16x16x32_f16(false, (A), false, (B), (short)0, (C), false, false)

// ---------------- prep: convert shared weights to f16, combine biases ----------------
__global__ void k_prep_const(const float* emb, const float* wihf, const float* whhf,
                             const float* bihf, const float* bhhf,
                             const float* wihb, const float* whhb,
                             const float* bihb, const float* bhhb,
                             const float* w1, char* ws) {
  int i = blockIdx.x * 256 + threadIdx.x;   // grid covers 16384
  if (i < VOCAB * EDIM) ((gf16*)g_u64(ws + OFF_EMB))[i]   = (f16)emb[i];
  if (i < G4H * EDIM)   ((gf16*)g_u64(ws + OFF_WIH_F))[i] = (f16)wihf[i];
  if (i < G4H * HDIM)   ((gf16*)g_u64(ws + OFF_WHH_F))[i] = (f16)whhf[i];
  if (i < G4H * EDIM)   ((gf16*)g_u64(ws + OFF_WIH_B))[i] = (f16)wihb[i];
  if (i < G4H * HDIM)   ((gf16*)g_u64(ws + OFF_WHH_B))[i] = (f16)whhb[i];
  if (i < 32 * 128)     ((gf16*)g_u64(ws + OFF_W1))[i]    = (f16)w1[i];
  if (i < G4H) {
    ((gf32*)g_u64(ws + OFF_BIAS_F))[i] = bihf[i] + bhhf[i];
    ((gf32*)g_u64(ws + OFF_BIAS_B))[i] = bihb[i] + bhhb[i];
  }
}

// per-half GAT weights -> padded f16 [HP][HP]
__global__ void k_prep_gat(const float* wl, const float* wr, char* ws) {
  int i = blockIdx.x * 256 + threadIdx.x;
  int tot = HP * HP;
  if (i >= 2 * tot) return;
  const gf32* w = (const gf32*)g_u64((i < tot) ? wl : wr);
  gf16* dst = (gf16*)g_u64(ws + ((i < tot) ? OFF_WL : OFF_WR));
  int j = (i < tot) ? i : i - tot;
  int r = j / HP, c = j % HP;
  dst[j] = (r < HID && c < HID) ? (f16)w[r * HID + c] : (f16)0.f;
}

// one gate pre-activation (no bias): x@Wih^T + h@Whh^T; C starts as inline 0
__device__ __forceinline__ v8f gate_acc(const gf16* wih, const gf16* whh,
                                        int jb, int mrow, int hi,
                                        v16h ax, v16h ah0, v16h ah1) {
  v8f acc = WMMA_F16(ax,
                     *(const g_v16h*)(wih + (size_t)(jb + mrow) * EDIM + hi * 16),
                     (v8f){});
  acc = WMMA_F16(ah0,
                 *(const g_v16h*)(whh + (size_t)(jb + mrow) * HDIM + hi * 16),
                 acc);
  acc = WMMA_F16(ah1,
                 *(const g_v16h*)(whh + (size_t)(jb + mrow) * HDIM + 32 + hi * 16),
                 acc);
  return acc;
}

// ---------------- BiLSTM(last) + MLP per edge, WMMA 16x16x32 f16 ----------------
// one wave handles a tile of 16 edges; M = edge rows
__global__ __launch_bounds__(256) void k_lstm(const char* __restrict__ ws,
                                              const int* __restrict__ tokens,
                                              float* __restrict__ ea,
                                              const float* __restrict__ b1,
                                              const float* __restrict__ w2,
                                              const float* __restrict__ b2) {
  __shared__ f16 sEmb[VOCAB * EDIM];        // 8 KB
  __shared__ f16 sHT[8][16 * 128];          // 32 KB: per-wave transpose / MLP staging
  for (int i = threadIdx.x; i < VOCAB * EDIM; i += 256)
    sEmb[i] = ((const gf16*)g_u64(ws + OFF_EMB))[i];
  __syncthreads();

  const int wid  = threadIdx.x >> 5;
  const int lane = threadIdx.x & 31;
  const int tile = blockIdx.x * 8 + wid;
  if (tile * 16 >= N_EDGES) return;
  const int ebase = tile * 16;
  const int mrow  = lane & 15;              // edge row (A) / output column (D)
  const int hi    = lane >> 4;              // half-wave index
  const int e_row = ebase + mrow;

  const gi32* tokg = (const gi32*)g_u64(tokens);
  f16* ht = &sHT[wid][0];                   // row stride 128 halves

  // preload per-lane gate biases into registers (immune to the asm memory clobber)
  float bI[4], bF[4], bG[4], bO[4];
  {
    const gf32* bias = (const gf32*)g_u64(ws + OFF_BIAS_F);
    #pragma unroll
    for (int tt = 0; tt < 4; ++tt) {
      bI[tt] = bias[0 * HDIM + tt * 16 + mrow];
      bF[tt] = bias[1 * HDIM + tt * 16 + mrow];
      bG[tt] = bias[2 * HDIM + tt * 16 + mrow];
      bO[tt] = bias[3 * HDIM + tt * 16 + mrow];
    }
  }

  v8f c[4];
  #pragma unroll
  for (int t = 0; t < 4; ++t) c[t] = (v8f){};
  v16h ah0 = (v16h){}, ah1 = (v16h){};      // h==0 at step 0

  #pragma unroll 1
  for (int t = 0; t < SEQ; ++t) {
    // opaque global bases: re-materialize weight loads each step (no hoist+spill),
    // and keep them on the global_load path (LOADcnt only, no DScnt coupling)
    const gf16* wih = (const gf16*)g_u64_opq(ws + OFF_WIH_F);
    const gf16* whh = (const gf16*)g_u64_opq(ws + OFF_WHH_F);

    int tok = tokg[e_row * SEQ + t];
    int kb = hi * 16;
    v16h ax = cat16(*(const v8h*)&sEmb[tok * EDIM + kb],
                    *(const v8h*)&sEmb[tok * EDIM + kb + 8]);
    #pragma unroll
    for (int tt = 0; tt < 4; ++tt) {
      // sequential gates: only one accumulator + one temp live at a time
      v8f gi = gate_acc(wih, whh, 0 * HDIM + tt * 16, mrow, hi, ax, ah0, ah1);
      v8f si;
      #pragma unroll
      for (int v = 0; v < 8; ++v) si[v] = sigm(gi[v] + bI[tt]);
      v8f gf = gate_acc(wih, whh, 1 * HDIM + tt * 16, mrow, hi, ax, ah0, ah1);
      #pragma unroll
      for (int v = 0; v < 8; ++v) c[tt][v] = sigm(gf[v] + bF[tt]) * c[tt][v];
      v8f gg = gate_acc(wih, whh, 2 * HDIM + tt * 16, mrow, hi, ax, ah0, ah1);
      #pragma unroll
      for (int v = 0; v < 8; ++v) c[tt][v] += si[v] * tanh_fast(gg[v] + bG[tt]);
      v8f go = gate_acc(wih, whh, 3 * HDIM + tt * 16, mrow, hi, ax, ah0, ah1);
      #pragma unroll
      for (int v = 0; v < 8; ++v) {
        float hv = sigm(go[v] + bO[tt]) * tanh_fast(c[tt][v]);
        ht[(hi * 8 + v) * 128 + tt * 16 + mrow] = (f16)hv;  // D-layout -> row-major LDS
      }
    }
    asm volatile("s_wait_dscnt 0" ::: "memory");
    // reload h in A layout for the next step
    int kb0 = hi * 16;
    ah0 = cat16(*(const v8h*)&ht[mrow * 128 + kb0],
                *(const v8h*)&ht[mrow * 128 + kb0 + 8]);
    ah1 = cat16(*(const v8h*)&ht[mrow * 128 + 32 + kb0],
                *(const v8h*)&ht[mrow * 128 + 32 + kb0 + 8]);
  }

  // backward direction: single step on last token, h0=c0=0 (drop Whh terms)
  {
    const gf16* wihB  = (const gf16*)g_u64_opq(ws + OFF_WIH_B);
    const gf32* biasB = (const gf32*)g_u64(ws + OFF_BIAS_B);
    int tok = tokg[e_row * SEQ + (SEQ - 1)];
    int kb = hi * 16;
    v16h ax = cat16(*(const v8h*)&sEmb[tok * EDIM + kb],
                    *(const v8h*)&sEmb[tok * EDIM + kb + 8]);
    #pragma unroll
    for (int tt = 0; tt < 4; ++tt) {
      float cbI = biasB[0 * HDIM + tt * 16 + mrow];
      float cbG = biasB[2 * HDIM + tt * 16 + mrow];
      float cbO = biasB[3 * HDIM + tt * 16 + mrow];
      v8f gi = WMMA_F16(ax,
          *(const g_v16h*)(wihB + (size_t)(0 * HDIM + tt * 16 + mrow) * EDIM + hi * 16),
          (v8f){});
      v8f gg = WMMA_F16(ax,
          *(const g_v16h*)(wihB + (size_t)(2 * HDIM + tt * 16 + mrow) * EDIM + hi * 16),
          (v8f){});
      v8f go = WMMA_F16(ax,
          *(const g_v16h*)(wihB + (size_t)(3 * HDIM + tt * 16 + mrow) * EDIM + hi * 16),
          (v8f){});
      #pragma unroll
      for (int v = 0; v < 8; ++v) {
        float cv = sigm(gi[v] + cbI) * tanh_fast(gg[v] + cbG);
        float hb = sigm(go[v] + cbO) * tanh_fast(cv);
        ht[(hi * 8 + v) * 128 + 64 + tt * 16 + mrow] = (f16)hb;   // cols 64..127
      }
    }
  }
  asm volatile("s_wait_dscnt 0" ::: "memory");

  // MLP: z = relu([h_fwd,h_bwd] @ W1^T + b1) (16x32), ea = relu(z @ W2^T + b2)
  const gf16* w1h = (const gf16*)g_u64(ws + OFF_W1);
  v8f z0 = (v8f){}, z1 = (v8f){};
  #pragma unroll
  for (int kc = 0; kc < 4; ++kc) {
    int kb = kc * 32 + hi * 16;
    v16h a = cat16(*(const v8h*)&ht[mrow * 128 + kb],
                   *(const v8h*)&ht[mrow * 128 + kb + 8]);
    v16h bA = *(const g_v16h*)(w1h + (size_t)(mrow) * 128 + kc * 32 + hi * 16);
    z0 = WMMA_F16(a, bA, z0);
    v16h bB = *(const g_v16h*)(w1h + (size_t)(16 + mrow) * 128 + kc * 32 + hi * 16);
    z1 = WMMA_F16(a, bB, z1);
  }
  float b0 = b1[mrow], b16 = b1[16 + mrow];
  float w2a = w2[mrow], w2b = w2[16 + mrow];
  float part[8];
  #pragma unroll
  for (int v = 0; v < 8; ++v)
    part[v] = fmaxf(z0[v] + b0, 0.f) * w2a + fmaxf(z1[v] + b16, 0.f) * w2b;
  #pragma unroll
  for (int off = 1; off < 16; off <<= 1)
    #pragma unroll
    for (int v = 0; v < 8; ++v)
      part[v] += __shfl_xor(part[v], off, 16);
  float myv = 0.f;
  #pragma unroll
  for (int v = 0; v < 8; ++v) if (mrow == v) myv = part[v];
  if (mrow < 8)
    ((gf32*)g_u64(ea))[ebase + hi * 8 + mrow] = fmaxf(myv + b2[0], 0.f);
}

// ---------------- x copy into d_out half ----------------
__global__ void k_copy_x(const float* __restrict__ x, float* __restrict__ out, int off) {
  long long i = (long long)blockIdx.x * 256 + threadIdx.x;
  if (i >= (long long)N_NODES * HID) return;
  int n = (int)(i / HID), d = (int)(i % HID);
  ((gf32*)g_u64(out))[(size_t)n * OUTC + off + d] = ((const gf32*)g_u64(x))[i];
}

// scatter: out[row, off+REGEX+argmax(x[dst,:53])] = ea[e]
__global__ void k_scatter(const float* __restrict__ x, const int* __restrict__ ei,
                          const float* __restrict__ ea, float* __restrict__ out,
                          int off, int write_src) {
  int e = blockIdx.x * 256 + threadIdx.x;
  if (e >= N_EDGES) return;
  const gi32* eig = (const gi32*)g_u64(ei);
  int src = eig[e], dst = eig[N_EDGES + e];
  const gf32* xr = (const gf32*)g_u64(x) + (size_t)dst * HID;
  float best = xr[0]; int bi = 0;
  for (int d = 1; d < STATE; ++d) { float v = xr[d]; if (v > best) { best = v; bi = d; } }
  int row = write_src ? src : dst;
  ((gf32*)g_u64(out))[(size_t)row * OUTC + off + REGEX + bi] = ((const gf32*)g_u64(ea))[e];
}

// build padded f16 xh from d_out half; zero acc; init m/den
__global__ void k_xh_init(const float* __restrict__ out, int off, char* ws) {
  long long i = (long long)blockIdx.x * 256 + threadIdx.x;
  if (i >= (long long)N_NODES * HP) return;
  int n = (int)(i / HP), d = (int)(i % HP);
  ((gf16*)g_u64(ws + OFF_XH))[i] =
      (d < HID) ? (f16)((const gf32*)g_u64(out))[(size_t)n * OUTC + off + d] : (f16)0.f;
  ((gf32*)g_u64(ws + OFF_ACC))[i] = 0.f;
  if (i < N_NODES) {
    ((gu32*)g_u64(ws + OFF_M))[i] = 0u;     // ordered encoding of -inf-ish
    ((gf32*)g_u64(ws + OFF_DEN))[i] = 0.f;
  }
}

// xl = xh @ Wl^T + bl ; xr = xh @ Wr^T + br   (WMMA, K padded to 224)
__global__ __launch_bounds__(256) void k_gemm(const char* __restrict__ ws,
                                              const float* __restrict__ bl,
                                              const float* __restrict__ br) {
  int wid = threadIdx.x >> 5, lane = threadIdx.x & 31;
  int rt = blockIdx.x * 8 + wid;
  if (rt >= N_NODES / 16) return;           // 6250 row tiles
  int ct = blockIdx.y, lr = blockIdx.z;
  const gf16* xh = (const gf16*)g_u64(ws + OFF_XH);
  const gf16* w  = (const gf16*)g_u64(ws + (lr ? OFF_WR : OFF_WL));
  gf32* y        = (gf32*)g_u64(ws + (lr ? OFF_XR : OFF_XL));
  const gf32* bias = (const gf32*)g_u64(lr ? br : bl);

  int mrow = lane & 15, hi = lane >> 4;
  int col = ct * 16 + mrow;
  float bv = (col < HID) ? bias[col] : 0.f;
  v8f acc = (v8f){};

  const gf16* ap = xh + (size_t)(rt * 16 + mrow) * HP;
  const gf16* bp = w + (size_t)col * HP;
  #pragma unroll
  for (int kc = 0; kc < 7; ++kc) {
    int ka = kc * 32 + hi * 16;
    v16h a = cat16(*(const g_v8h*)(ap + ka), *(const g_v8h*)(ap + ka + 8));
    v16h b = *(const g_v16h*)(bp + kc * 32 + hi * 16);
    acc = WMMA_F16(a, b, acc);
  }
  if (col < HID) {
    #pragma unroll
    for (int v = 0; v < 8; ++v)
      y[(size_t)(rt * 16 + hi * 8 + v) * HP + col] = acc[v] + bv;
  }
}

// e = leaky(xl[src]+xr[dst]) . att ; segment max via ordered atomicMax
__global__ __launch_bounds__(256) void k_edge_e(char* __restrict__ ws,
                                                const int* __restrict__ ei,
                                                const float* __restrict__ att) {
  int wid = threadIdx.x >> 5, lane = threadIdx.x & 31;
  int e = blockIdx.x * 8 + wid;
  if (e >= N_EDGES) return;
  const gi32* eig = (const gi32*)g_u64(ei);
  int src = eig[e], dst = eig[N_EDGES + e];
  const gf32* xl = (const gf32*)g_u64(ws + OFF_XL) + (size_t)src * HP;
  const gf32* xr = (const gf32*)g_u64(ws + OFF_XR) + (size_t)dst * HP;
  const gf32* attg = (const gf32*)g_u64(att);
  float s = 0.f;
  for (int d = lane; d < HID; d += 32) {
    float u = xl[d] + xr[d];
    u = (u > 0.f) ? u : NEG * u;
    s += u * attg[d];
  }
  #pragma unroll
  for (int off = 16; off; off >>= 1) s += __shfl_xor(s, off, 32);
  if (lane == 0) {
    ((gf32*)g_u64(ws + OFF_ER))[e] = s;
    atomicMax((unsigned*)(ws + OFF_M) + dst, f2ord(s));
  }
}

// ex = exp(e - m[dst]); denom via atomicAdd
__global__ void k_edge_ex(char* __restrict__ ws, const int* __restrict__ ei) {
  int e = blockIdx.x * 256 + threadIdx.x;
  if (e >= N_EDGES) return;
  int dst = ((const gi32*)g_u64(ei))[N_EDGES + e];
  float m = ord2f(((const gu32*)g_u64(ws + OFF_M))[dst]);
  float v = __expf(((const gf32*)g_u64(ws + OFF_ER))[e] - m);
  ((gf32*)g_u64(ws + OFF_EX))[e] = v;
  atomicAdd((float*)(ws + OFF_DEN) + dst, v);
}

// acc[dst] += alpha * xl[src]
__global__ __launch_bounds__(256) void k_edge_acc(char* __restrict__ ws,
                                                  const int* __restrict__ ei) {
  int wid = threadIdx.x >> 5, lane = threadIdx.x & 31;
  int e = blockIdx.x * 8 + wid;
  if (e >= N_EDGES) return;
  const gi32* eig = (const gi32*)g_u64(ei);
  int src = eig[e], dst = eig[N_EDGES + e];
  float alpha = ((const gf32*)g_u64(ws + OFF_EX))[e] /
                ((const gf32*)g_u64(ws + OFF_DEN))[dst];
  const gf32* xl = (const gf32*)g_u64(ws + OFF_XL) + (size_t)src * HP;
  float* acc = (float*)(ws + OFF_ACC) + (size_t)dst * HP;
  for (int d = lane; d < HID; d += 32)
    atomicAdd(acc + d, alpha * xl[d]);
}

// out = relu(acc + gbias + x)
__global__ void k_final(const char* __restrict__ ws, const float* __restrict__ gbias,
                        float* __restrict__ out, int off) {
  long long i = (long long)blockIdx.x * 256 + threadIdx.x;
  if (i >= (long long)N_NODES * HID) return;
  int n = (int)(i / HID), d = (int)(i % HID);
  size_t o = (size_t)n * OUTC + off + d;
  gf32* og = (gf32*)g_u64(out);
  float v = ((const gf32*)g_u64(ws + OFF_ACC))[(size_t)n * HP + d] +
            ((const gf32*)g_u64(gbias))[d] + og[o];
  og[o] = fmaxf(v, 0.f);
}

// ---------------- host launcher ----------------
extern "C" void kernel_launch(void* const* d_in, const int* in_sizes, int n_in,
                              void* d_out, int out_size, void* d_ws, size_t ws_size,
                              hipStream_t stream) {
  (void)in_sizes; (void)n_in; (void)out_size; (void)ws_size;
  const float* x_in[2]  = { (const float*)d_in[0], (const float*)d_in[1] };
  const float* emb      = (const float*)d_in[2];
  const float* wihf = (const float*)d_in[3],  *whhf = (const float*)d_in[4];
  const float* bihf = (const float*)d_in[5],  *bhhf = (const float*)d_in[6];
  const float* wihb = (const float*)d_in[7],  *whhb = (const float*)d_in[8];
  const float* bihb = (const float*)d_in[9],  *bhhb = (const float*)d_in[10];
  const float* w1 = (const float*)d_in[11], *b1 = (const float*)d_in[12];
  const float* w2 = (const float*)d_in[13], *b2 = (const float*)d_in[14];
  const float* gWl[2]   = { (const float*)d_in[15], (const float*)d_in[21] };
  const float* gbl[2]   = { (const float*)d_in[16], (const float*)d_in[22] };
  const float* gWr[2]   = { (const float*)d_in[17], (const float*)d_in[23] };
  const float* gbr[2]   = { (const float*)d_in[18], (const float*)d_in[24] };
  const float* gatt[2]  = { (const float*)d_in[19], (const float*)d_in[25] };
  const float* gbias[2] = { (const float*)d_in[20], (const float*)d_in[26] };
  const int* ei[2]  = { (const int*)d_in[27], (const int*)d_in[28] };
  const int* tok[2] = { (const int*)d_in[29], (const int*)d_in[30] };

  char* ws = (char*)d_ws;
  float* out = (float*)d_out;
  float* ea = (float*)(ws + OFF_EA);

  k_prep_const<<<64, 256, 0, stream>>>(emb, wihf, whhf, bihf, bhhf,
                                       wihb, whhb, bihb, bhhb, w1, ws);

  for (int hv = 0; hv < 2; ++hv) {
    int off = hv ? HID : 0;
    int write_src = hv ? 0 : 1;

    k_prep_gat<<<(2 * HP * HP + 255) / 256, 256, 0, stream>>>(gWl[hv], gWr[hv], ws);
    k_lstm<<<N_EDGES / (16 * 8), 256, 0, stream>>>(ws, tok[hv], ea, b1, w2, b2);
    k_copy_x<<<(int)(((long long)N_NODES * HID + 255) / 256), 256, 0, stream>>>(x_in[hv], out, off);
    k_scatter<<<(N_EDGES + 255) / 256, 256, 0, stream>>>(x_in[hv], ei[hv], ea, out, off, write_src);
    k_xh_init<<<(int)(((long long)N_NODES * HP + 255) / 256), 256, 0, stream>>>(out, off, ws);
    dim3 gg((N_NODES / 16 + 7) / 8, HP / 16, 2);
    k_gemm<<<gg, 256, 0, stream>>>(ws, gbl[hv], gbr[hv]);
    k_edge_e<<<N_EDGES / 8, 256, 0, stream>>>(ws, ei[hv], gatt[hv]);
    k_edge_ex<<<(N_EDGES + 255) / 256, 256, 0, stream>>>(ws, ei[hv]);
    k_edge_acc<<<N_EDGES / 8, 256, 0, stream>>>(ws, ei[hv]);
    k_final<<<(int)(((long long)N_NODES * HID + 255) / 256), 256, 0, stream>>>(ws, gbias[hv], out, off);
  }
}
